// QuantumFundusClassifier_74053826118342
// MI455X (gfx1250) — matrix operand order repaired and checked by reading, not compiled, not run on hardware
//
#include <hip/hip_runtime.h>

typedef _Float16 v16h __attribute__((ext_vector_type(16)));
typedef _Float16 v8h  __attribute__((ext_vector_type(8)));
typedef _Float16 v4h  __attribute__((ext_vector_type(4)));
typedef float    v8f  __attribute__((ext_vector_type(8)));

#define NROWS   16384
#define KDIM    1280
#define NDIM    64
#define MBLK    128
#define KC      32
#define NCHUNK  (KDIM / KC)    // 40
#define ASTR    40             // padded stride (halves) for A rows in LDS
#define BSTR    40             // padded stride (halves) for B rows in LDS
#define HSTR    68             // padded stride (floats) for h tile in LDS
#define BUFH    15360          // halves per double-buffer slot (30720 B)

__global__ __launch_bounds__(256)
void qfc_fused(const float* __restrict__ x,
               const float* __restrict__ W1,
               const float* __restrict__ b1,
               const float* __restrict__ W2,
               const float* __restrict__ b2,
               const float* __restrict__ qw,
               const float* __restrict__ Wc1,
               const float* __restrict__ bc1,
               const float* __restrict__ Wc2,
               const float* __restrict__ bc2,
               float* __restrict__ out)
{
  __shared__ __align__(16) unsigned char smem[61440];   // 2 x 30720B buffers; H tile reuses
  _Float16* base = (_Float16*)smem;

  const int tid   = threadIdx.x;
  const int wave  = tid >> 5;
  const int lane  = tid & 31;
  const int halfg = lane >> 4;   // K-half selector per WMMA layout
  const int l16   = lane & 15;   // M row (A) / N col (B,C)
  const int row0  = blockIdx.x * MBLK;

  v8f acc[4];
  #pragma unroll
  for (int nt=0; nt<4; nt++)
    #pragma unroll
    for (int r=0; r<8; r++) acc[nt][r] = 0.f;

  // register staging for next chunk's global data
  float4 xa[4];
  float4 wb[2];

  auto global_stage = [&](int ch){
    const int k0 = ch*KC;
    #pragma unroll
    for (int i=0; i<4; i++){
      int j  = tid + 256*i;          // 0..1023 float4 slots of x tile [128][32]
      int r  = j >> 3;
      int kq = j & 7;
      xa[i] = *(const float4*)(x + (size_t)(row0 + r)*KDIM + k0 + kq*4);
    }
    #pragma unroll
    for (int i=0; i<2; i++){
      int j  = tid + 256*i;          // 0..511 float4 slots of W1 chunk [32][64]
      int k  = j >> 4;
      int nq = j & 15;
      wb[i] = *(const float4*)(W1 + (size_t)(k0 + k)*NDIM + nq*4);
    }
  };

  auto lds_store = [&](int buf){
    _Float16* Ah = base + buf*BUFH;
    _Float16* Al = Ah + MBLK*ASTR;     // +5120 halves
    _Float16* Bh = Al + MBLK*ASTR;     // +10240
    _Float16* Bl = Bh + NDIM*BSTR;     // +12800
    #pragma unroll
    for (int i=0; i<4; i++){
      int j  = tid + 256*i;
      int r  = j >> 3;
      int kq = j & 7;
      float f[4] = {xa[i].x, xa[i].y, xa[i].z, xa[i].w};
      v4h hi, lo;
      #pragma unroll
      for (int e=0; e<4; e++){
        _Float16 h = (_Float16)f[e];
        hi[e] = h;
        lo[e] = (_Float16)(f[e] - (float)h);
      }
      *(v4h*)(Ah + r*ASTR + kq*4) = hi;
      *(v4h*)(Al + r*ASTR + kq*4) = lo;
    }
    #pragma unroll
    for (int i=0; i<2; i++){
      int j  = tid + 256*i;
      int k  = j >> 4;
      int nq = j & 15;
      float f[4] = {wb[i].x, wb[i].y, wb[i].z, wb[i].w};
      #pragma unroll
      for (int e=0; e<4; e++){
        _Float16 h = (_Float16)f[e];
        Bh[(nq*4+e)*BSTR + k] = h;
        Bl[(nq*4+e)*BSTR + k] = (_Float16)(f[e] - (float)h);
      }
    }
  };

  auto compute = [&](int buf){
    const _Float16* Ah = base + buf*BUFH;
    const _Float16* Al = Ah + MBLK*ASTR;
    const _Float16* Bh = Al + MBLK*ASTR;
    const _Float16* Bl = Bh + NDIM*BSTR;

    // load ALL fragments first -> one DS clause, then stream 12 WMMAs
    const _Float16* arow_h = Ah + (wave*16 + l16)*ASTR;
    const _Float16* arow_l = Al + (wave*16 + l16)*ASTR;
    v16h a_hi, a_lo;
    {
      v8h t0 = *(const v8h*)(arow_h + halfg*8);        // K = half*8 + 0..7
      v8h t1 = *(const v8h*)(arow_h + 16 + halfg*8);   // K = 16 + half*8 + 0..7
      v8h u0 = *(const v8h*)(arow_l + halfg*8);
      v8h u1 = *(const v8h*)(arow_l + 16 + halfg*8);
      #pragma unroll
      for (int e=0; e<8; e++){ a_hi[e]=t0[e]; a_hi[e+8]=t1[e]; a_lo[e]=u0[e]; a_lo[e+8]=u1[e]; }
    }
    v16h b_hi[4], b_lo[4];
    #pragma unroll
    for (int nt=0; nt<4; nt++){
      const _Float16* brow_h = Bh + (nt*16 + l16)*BSTR + halfg*8;
      const _Float16* brow_l = Bl + (nt*16 + l16)*BSTR + halfg*8;
      v8h t0 = *(const v8h*)(brow_h);      // K = half*8 + 0..7
      v8h t1 = *(const v8h*)(brow_h + 16); // K = 16 + half*8 + 0..7
      v8h u0 = *(const v8h*)(brow_l);
      v8h u1 = *(const v8h*)(brow_l + 16);
      #pragma unroll
      for (int e=0; e<8; e++){
        b_hi[nt][e]=t0[e]; b_hi[nt][e+8]=t1[e];
        b_lo[nt][e]=u0[e]; b_lo[nt][e+8]=u1[e];
      }
    }
    #pragma unroll
    for (int nt=0; nt<4; nt++){
      // split-precision: hi*hi + lo*hi + hi*lo  (~fp32 accuracy)
      acc[nt] = __builtin_amdgcn_wmma_f32_16x16x32_f16(false, a_hi, false, b_hi[nt], (short)0, acc[nt], false, false);
      acc[nt] = __builtin_amdgcn_wmma_f32_16x16x32_f16(false, a_lo, false, b_hi[nt], (short)0, acc[nt], false, false);
      acc[nt] = __builtin_amdgcn_wmma_f32_16x16x32_f16(false, a_hi, false, b_lo[nt], (short)0, acc[nt], false, false);
    }
  };

  // ---- software-pipelined main loop: one barrier per iteration ----
  global_stage(0);
  lds_store(0);
  __syncthreads();
  for (int ch=0; ch<NCHUNK; ch++){
    const int cur = ch & 1;
    if (ch+1 < NCHUNK) global_stage(ch+1);   // issue next chunk's global loads early
    compute(cur);                            // DS clause + 12 WMMAs on current buffer
    if (ch+1 < NCHUNK) lds_store(1-cur);     // fill other buffer (last read in previous iter)
    __syncthreads();
  }

  // ---- bias + relu, scatter h tile to LDS per C/D layout ----
  float* Hs = (float*)smem;   // [128][HSTR]
  #pragma unroll
  for (int nt=0; nt<4; nt++){
    int col = nt*16 + l16;
    float bias = b1[col];
    #pragma unroll
    for (int r=0; r<8; r++){
      int row = wave*16 + halfg*8 + r;
      Hs[row*HSTR + col] = fmaxf(acc[nt][r] + bias, 0.f);
    }
  }
  __syncthreads();

  // ---- stage 2: per-row 64->4 tanh, 4-qubit sim, 8->16->1 head ----
  if (tid < MBLK) {
    const float* hrow = Hs + tid*HSTR;
    float red[4];
    #pragma unroll
    for (int j=0; j<4; j++){
      float s = b2[j];
      for (int i=0; i<NDIM; i++) s += hrow[i]*W2[i*4+j];
      red[j] = tanhf(s);
    }
    // statevector: 16 complex amps; qubit w <-> bit (3-w)
    float pr[16], pi[16];
    #pragma unroll
    for (int i=0; i<16; i++){ pr[i]=0.f; pi[i]=0.f; }
    pr[0] = 1.f;

    auto RY = [&](int mask, float t){
      float c = cosf(0.5f*t), s = sinf(0.5f*t);
      #pragma unroll
      for (int i=0; i<16; i++) if (!(i & mask)){
        int j = i | mask;
        float r0=pr[i], i0=pi[i], r1=pr[j], i1=pi[j];
        pr[i]=c*r0 - s*r1; pi[i]=c*i0 - s*i1;
        pr[j]=s*r0 + c*r1; pi[j]=s*i0 + c*i1;
      }
    };
    auto RZ = [&](int mask, float t){
      float c = cosf(0.5f*t), s = sinf(0.5f*t);
      #pragma unroll
      for (int i=0; i<16; i++){
        float r=pr[i], im=pi[i];
        if (!(i & mask)) { pr[i]=c*r + s*im; pi[i]=c*im - s*r; }   // * (c - i s)
        else             { pr[i]=c*r - s*im; pi[i]=c*im + s*r; }   // * (c + i s)
      }
    };
    auto CNOT = [&](int cmask, int tmask){
      #pragma unroll
      for (int i=0; i<16; i++) if ((i & cmask) && !(i & tmask)){
        int j = i | tmask;
        float tr=pr[i], ti=pi[i];
        pr[i]=pr[j]; pi[i]=pi[j];
        pr[j]=tr;    pi[j]=ti;
      }
    };

    RY(8, red[0]); RY(4, red[1]); RY(2, red[2]); RY(1, red[3]);
    #pragma unroll
    for (int l=0; l<3; l++){
      RY(8, qw[(l*4+0)*2+0]); RZ(8, qw[(l*4+0)*2+1]);
      RY(4, qw[(l*4+1)*2+0]); RZ(4, qw[(l*4+1)*2+1]);
      RY(2, qw[(l*4+2)*2+0]); RZ(2, qw[(l*4+2)*2+1]);
      RY(1, qw[(l*4+3)*2+0]); RZ(1, qw[(l*4+3)*2+1]);
      CNOT(8,4); CNOT(4,2); CNOT(2,1); CNOT(1,8);
    }
    float qout[4];
    #pragma unroll
    for (int w=0; w<4; w++){
      int mask = 8 >> w;
      float s = 0.f;
      #pragma unroll
      for (int i=0; i<16; i++){
        float p = pr[i]*pr[i] + pi[i]*pi[i];
        s += (i & mask) ? -p : p;
      }
      qout[w] = s;
    }
    float comb[8] = {red[0],red[1],red[2],red[3],qout[0],qout[1],qout[2],qout[3]};
    float logit = bc2[0];
    #pragma unroll
    for (int k=0; k<16; k++){
      float a = bc1[k];
      #pragma unroll
      for (int j=0; j<8; j++) a += comb[j]*Wc1[j*16+k];
      logit += fmaxf(a, 0.f)*Wc2[k];
    }
    out[row0 + tid] = logit;
  }
}

extern "C" void kernel_launch(void* const* d_in, const int* in_sizes, int n_in,
                              void* d_out, int out_size, void* d_ws, size_t ws_size,
                              hipStream_t stream) {
  const float* x   = (const float*)d_in[0];
  const float* W1  = (const float*)d_in[1];
  const float* b1  = (const float*)d_in[2];
  const float* W2  = (const float*)d_in[3];
  const float* b2  = (const float*)d_in[4];
  const float* qw  = (const float*)d_in[5];
  const float* Wc1 = (const float*)d_in[6];
  const float* bc1 = (const float*)d_in[7];
  const float* Wc2 = (const float*)d_in[8];
  const float* bc2 = (const float*)d_in[9];
  float* out = (float*)d_out;
  (void)in_sizes; (void)n_in; (void)out_size; (void)d_ws; (void)ws_size;
  qfc_fused<<<dim3(NROWS/MBLK), dim3(256), 0, stream>>>(x, W1, b1, W2, b2, qw, Wc1, bc1, Wc2, bc2, out);
}
